// EGNNDynamics_75144747810823
// MI455X (gfx1250) — matrix operand order repaired and checked by reading, not compiled, not run on hardware
//
#include <hip/hip_runtime.h>
#include <math.h>

// ---------------- problem constants ----------------
#define NB      4
#define NA_     24
#define NR_     176
#define NPB     200       // nodes (and edges per src) per batch
#define NTOTA   96
#define NTOTR   704
#define NTOT_   800
#define EALL    160000
#define EAT     2304
#define HID     64
#define PI_F    3.14159265358979323846f
#define BASEF   (1.0f/9.0f)

// ---------------- flat param offsets (jax pytree: sorted dict keys, W before b) -------------
// order: atom_dec, atom_enc, egnn{blocks[{coord,gcls}],emb,emb_out}, res_dec, res_enc
#define OFF_AD0_W 0        // [16][32]
#define OFF_AD0_B 512
#define OFF_AD1_W 544      // [32][16]
#define OFF_AD1_B 1056
#define OFF_AD2_W 1072     // [16][3]
#define OFF_AD2_B 1120
#define OFF_AE0_W 1123     // [16][32]
#define OFF_AE0_B 1635
#define OFF_AE1_W 1667     // [32][16]
#define OFF_AE1_B 2179
#define OFF_BLOCKS 2195
#define BLOCK_STRIDE 65601
// within a block: coord first
//   c0_W +0 ([146][64]) c0_b +9344 ; c1_W +9408 ([64][64]) c1_b +13504 ; c2_W +13568 ([64][1]) c2_b +13632
#define GCLS_OFF 13633
#define GCL_STRIDE 25984
//   e0_W +0 ([146][64]) e0_b +9344 ; e1_W +9408 e1_b +13504 ; n0_W +13568 ([128][64]) n0_b +21760 ; n1_W +21824 n1_b +25920
#define OFF_EMB_W    264599   // [17][64]
#define OFF_EMB_B    265687
#define OFF_EMBO_W   265751   // [64][17]
#define OFF_EMBO_B   266839
#define OFF_RE0_W    268473   // [21][42]
#define OFF_RE0_B    269355
#define OFF_RE1_W    269397   // [42][16]
#define OFF_RE1_B    270069
#define NPARAM_TOT   270085

// ---------------- WMMA types ----------------
typedef __attribute__((ext_vector_type(16))) _Float16 v16h;
typedef __attribute__((ext_vector_type(8)))  float    v8f;

#define MT    13       // M tiles (208 padded rows of 200 edges)
#define MPAD  208
#define LDA   72       // A tile LDS stride (halfs) -- skew to dodge bank conflicts
#define LDBW  72       // B (weight) LDS stride (halfs)

// fast silu: x * v_rcp_f32(1 + exp(-x)) -- avoids the IEEE div expansion (~10 VALU ops -> 1 TRANS)
__device__ __forceinline__ float siluf(float x) {
  return x * __builtin_amdgcn_rcpf(1.0f + __expf(-x));
}
__device__ __forceinline__ int batch_of(int i) { return (i < NTOTA) ? (i / NA_) : ((i - NTOTA) / NR_); }
__device__ __forceinline__ int dst_of(int b, int k) { return (k < NA_) ? (b * NA_ + k) : (NTOTA + b * NR_ + (k - NA_)); }

// A fragment: 16x32 f16.  lanes 0-15: M=lane, K in {0..7}U{16..23}; lanes 16-31: same rows, K+8.
__device__ __forceinline__ v16h packA(const _Float16* At, int mi, int kk, int lane) {
  v16h a;
  int row = mi * 16 + (lane & 15);
  int hi = lane >> 4;
  const _Float16* base = At + row * LDA + kk * 32 + hi * 8;
#pragma unroll
  for (int i = 0; i < 16; ++i) {
    int k = (i < 8) ? i : (i + 8);
    a[i] = base[k];
  }
  return a;
}
// B fragment: 32x16 f16 (KxN). lanes 0-15: N=lane, K in {0..7}U{16..23}; lanes 16-31: N=lane-16, K+8.
__device__ __forceinline__ v16h packB(const _Float16* Wl, int kk, int ni, int lane) {
  v16h b;
  int n = ni * 16 + (lane & 15);
  int hi = lane >> 4;
  const _Float16* base = Wl + (kk * 32 + hi * 8) * LDBW + n;
#pragma unroll
  for (int i = 0; i < 16; ++i) b[i] = base[((i < 8) ? i : (i + 8)) * LDBW];
  return b;
}

// ---------------- node setup: encoders + pos_enc + [h, t] ----------------
__global__ void setup_nodes_kernel(const float* __restrict__ xa, const float* __restrict__ xr,
                                   const float* __restrict__ tsc, const float* __restrict__ P,
                                   float* __restrict__ H17, float* __restrict__ X, float* __restrict__ X0) {
  int i = blockIdx.x * blockDim.x + threadIdx.x;
  if (i >= NTOT_) return;
  float h[16];
  float tmp[42];
  float xyz[3];
  if (i < NTOTA) {
    const float* row = xa + i * 19;
    xyz[0] = row[0]; xyz[1] = row[1]; xyz[2] = row[2];
    const float* W0 = P + OFF_AE0_W; const float* b0 = P + OFF_AE0_B;
    const float* W1 = P + OFF_AE1_W; const float* b1 = P + OFF_AE1_B;
    for (int o = 0; o < 32; ++o) {
      float s = b0[o];
      for (int f = 0; f < 16; ++f) s += row[3 + f] * W0[f * 32 + o];
      tmp[o] = siluf(s);
    }
    for (int o = 0; o < 16; ++o) {
      float s = b1[o];
      for (int f = 0; f < 32; ++f) s += tmp[f] * W1[f * 16 + o];
      h[o] = s;
    }
    // positional encoding of index within batch
    int pos = i % NA_;
    float xv = (float)pos * PI_F;
    for (int j = 0; j < 8; ++j) {
      float ang = xv * (BASEF * (float)(1 << j));
      h[j]     += sinf(ang);
      h[8 + j] += cosf(ang);
    }
  } else {
    int r = i - NTOTA;
    const float* row = xr + r * 24;
    xyz[0] = row[0]; xyz[1] = row[1]; xyz[2] = row[2];
    const float* W0 = P + OFF_RE0_W; const float* b0 = P + OFF_RE0_B;
    const float* W1 = P + OFF_RE1_W; const float* b1 = P + OFF_RE1_B;
    for (int o = 0; o < 42; ++o) {
      float s = b0[o];
      for (int f = 0; f < 21; ++f) s += row[3 + f] * W0[f * 42 + o];
      tmp[o] = siluf(s);
    }
    for (int o = 0; o < 16; ++o) {
      float s = b1[o];
      for (int f = 0; f < 42; ++f) s += tmp[f] * W1[f * 16 + o];
      h[o] = s;
    }
  }
  for (int c = 0; c < 3; ++c) { X[i * 3 + c] = xyz[c]; X0[i * 3 + c] = xyz[c]; }
  for (int j = 0; j < 16; ++j) H17[i * 17 + j] = h[j];
  H17[i * 17 + 16] = tsc[0];
}

// ---------------- generic node linear: out = [resid+] [silu] (A(,A2) @ W + b) ----------------
__global__ void node_linear_kernel(const float* __restrict__ A, int din,
                                   const float* __restrict__ A2, int din2,
                                   const float* __restrict__ W, const float* __restrict__ bias,
                                   float* __restrict__ out, int dout, int n,
                                   int doSilu, int doResid) {
  int idx = blockIdx.x * blockDim.x + threadIdx.x;
  if (idx >= n * dout) return;
  int node = idx / dout, j = idx - node * dout;
  float acc = bias ? bias[j] : 0.0f;
  const float* a = A + node * din;
  for (int f = 0; f < din; ++f) acc += a[f] * W[f * dout + j];
  if (A2) {
    const float* a2 = A2 + node * din2;
    const float* W2 = W + din * dout;
    for (int f = 0; f < din2; ++f) acc += a2[f] * W2[f * dout + j];
  }
  if (doSilu) acc = siluf(acc);
  if (doResid) out[idx] += acc; else out[idx] = acc;
}

// ---------------- edge geometry: radial d and normalized diff (implicit edges) ----------------
__global__ void edge_geom_kernel(const float* __restrict__ X, float* __restrict__ dOut,
                                 float* __restrict__ cd) {
  int e = blockIdx.x * blockDim.x + threadIdx.x;
  if (e >= EALL) return;
  int src = e / NPB, k = e - src * NPB;
  int b = batch_of(src);
  int dst = dst_of(b, k);
  float dx = X[src * 3 + 0] - X[dst * 3 + 0];
  float dy = X[src * 3 + 1] - X[dst * 3 + 1];
  float dz = X[src * 3 + 2] - X[dst * 3 + 2];
  float r = dx * dx + dy * dy + dz * dz;
  dOut[e] = r;
  if (cd) {
    float inv = __builtin_amdgcn_rsqf(r + 1e-8f);   // v_rsq_f32
    cd[e * 3 + 0] = dx * inv;
    cd[e * 3 + 1] = dy * inv;
    cd[e * 3 + 2] = dz * inv;
  }
}

// ---------------- atom-pair positional edge attributes (16 dims) ----------------
__global__ void ea16_kernel(float* __restrict__ EA) {
  int e = blockIdx.x * blockDim.x + threadIdx.x;
  if (e >= EAT) return;
  int srcA = e / NA_, k = e - srcA * NA_;
  int b = srcA / NA_;
  int dstA = b * NA_ + k;
  float xv = (float)(srcA - dstA) * PI_F;
  for (int j = 0; j < 8; ++j) {
    float ang = xv * (BASEF * (float)(1 << j));
    EA[e * 16 + j]     = sinf(ang);
    EA[e * 16 + 8 + j] = cosf(ang);
  }
}

// ---------------- GCL edge stage: m = silu(silu(pre0) @ W1 + b1); AGG[src] = sum_e m /100 ----
__global__ void __launch_bounds__(256)
gcl_edge_kernel(const float* __restrict__ P1n, const float* __restrict__ P2n,
                const float* __restrict__ Dc, const float* __restrict__ D0,
                const float* __restrict__ EA, const float* __restrict__ Wea,
                const float* __restrict__ b0, const float* __restrict__ W1,
                const float* __restrict__ b1, float* __restrict__ AGG) {
  __shared__ _Float16 At[MPAD * LDA];
  __shared__ _Float16 Wl[64 * LDBW];
  __shared__ float Ps[64], B0s[64], B1s[64];
  __shared__ float WeaS[18 * 64];
  __shared__ float aggW[8][64];
  int tid = threadIdx.x, lane = tid & 31, wv = tid >> 5;
  int src = blockIdx.x;
  int b = batch_of(src);

  if (tid < 64) { Ps[tid] = P1n[src * 64 + tid]; B0s[tid] = b0[tid]; B1s[tid] = b1[tid]; }
  for (int i = tid; i < 18 * 64; i += 256) WeaS[i] = Wea[i];
  for (int i = tid; i < 64 * 64; i += 256) { int r = i >> 6, c = i & 63; Wl[r * LDBW + c] = (_Float16)W1[i]; }
  for (int i = tid; i < 8 * 64; i += 256) ((float*)aggW)[i] = 0.0f;
  __syncthreads();

  // build f16 A tile: A[k,f] = silu(Psrc + Pdst + d*w_d + d0*w_d0 + ea16 terms + b0)
  for (int ke = wv; ke < MPAD; ke += 8) {
    if (ke < NPB) {
      int dst = dst_of(b, ke);
      int e = src * NPB + ke;
      float d = Dc[e], dd0 = D0[e];
      bool atom = (src < NTOTA) && (ke < NA_);
      const float* eaR = EA + (src * NA_ + ke) * 16;  // only read when atom
      for (int f = lane; f < 64; f += 32) {
        float v = Ps[f] + P2n[dst * 64 + f] + d * WeaS[f] + dd0 * WeaS[64 + f] + B0s[f];
        if (atom) {
#pragma unroll
          for (int j = 0; j < 16; ++j) v += eaR[j] * WeaS[(2 + j) * 64 + f];
        }
        At[ke * LDA + f] = (_Float16)siluf(v);
      }
    } else {
      for (int f = lane; f < 64; f += 32) At[ke * LDA + f] = (_Float16)0.0f;
    }
  }
  __syncthreads();

  int hi = lane >> 4, n15 = lane & 15;
  for (int t = wv; t < MT * 4; t += 8) {     // ni = t&3 is wave-invariant -> B fragments hoisted
    int mi = t >> 2, ni = t & 3;
    v8f acc = {};
    v16h a0 = packA(At, mi, 0, lane);
    v16h bm0 = packB(Wl, 0, ni, lane);
    acc = __builtin_amdgcn_wmma_f32_16x16x32_f16(false, a0, false, bm0, (short)0, acc, false, false);
    v16h a1 = packA(At, mi, 1, lane);
    v16h bm1 = packB(Wl, 1, ni, lane);
    acc = __builtin_amdgcn_wmma_f32_16x16x32_f16(false, a1, false, bm1, (short)0, acc, false, false);
    int n = ni * 16 + n15;
    float bb = B1s[n];
    float csum = 0.0f;
#pragma unroll
    for (int r = 0; r < 8; ++r) {
      int row = mi * 16 + r + hi * 8;
      float m = siluf(acc[r] + bb);
      csum += (row < NPB) ? m : 0.0f;
    }
    float other = __shfl(csum, lane ^ 16, 32);
    if (lane < 16) aggW[wv][n] += csum + other;   // per-wave private, serial -> deterministic
  }
  __syncthreads();
  if (tid < 64) {
    float s = 0.0f;
    for (int w = 0; w < 8; ++w) s += aggW[w][tid];
    AGG[src * 64 + tid] = s * 0.01f;
  }
}

// ---------------- coord edge stage: phi = mlp3(...); X[src] += sum_e cdiff*phi / 100 --------
__global__ void __launch_bounds__(256)
coord_edge_kernel(const float* __restrict__ P1n, const float* __restrict__ P2n,
                  const float* __restrict__ Dc, const float* __restrict__ D0,
                  const float* __restrict__ EA, const float* __restrict__ CD,
                  const float* __restrict__ Wea, const float* __restrict__ b0,
                  const float* __restrict__ W1, const float* __restrict__ b1,
                  const float* __restrict__ w2, const float* __restrict__ b2,
                  float* __restrict__ X) {
  __shared__ _Float16 At[MPAD * LDA];
  __shared__ _Float16 Wl[64 * LDBW];
  __shared__ float Ps[64], B0s[64], B1s[64], W2s[64];
  __shared__ float WeaS[18 * 64];
  __shared__ float phiW[8][MPAD];
  __shared__ float contrib[NPB * 3];
  int tid = threadIdx.x, lane = tid & 31, wv = tid >> 5;
  int src = blockIdx.x;
  int b = batch_of(src);

  if (tid < 64) { Ps[tid] = P1n[src * 64 + tid]; B0s[tid] = b0[tid]; B1s[tid] = b1[tid]; W2s[tid] = w2[tid]; }
  for (int i = tid; i < 18 * 64; i += 256) WeaS[i] = Wea[i];
  for (int i = tid; i < 64 * 64; i += 256) { int r = i >> 6, c = i & 63; Wl[r * LDBW + c] = (_Float16)W1[i]; }
  for (int i = tid; i < 8 * MPAD; i += 256) ((float*)phiW)[i] = 0.0f;
  __syncthreads();

  for (int ke = wv; ke < MPAD; ke += 8) {
    if (ke < NPB) {
      int dst = dst_of(b, ke);
      int e = src * NPB + ke;
      float d = Dc[e], dd0 = D0[e];
      bool atom = (src < NTOTA) && (ke < NA_);
      const float* eaR = EA + (src * NA_ + ke) * 16;
      for (int f = lane; f < 64; f += 32) {
        float v = Ps[f] + P2n[dst * 64 + f] + d * WeaS[f] + dd0 * WeaS[64 + f] + B0s[f];
        if (atom) {
#pragma unroll
          for (int j = 0; j < 16; ++j) v += eaR[j] * WeaS[(2 + j) * 64 + f];
        }
        At[ke * LDA + f] = (_Float16)siluf(v);
      }
    } else {
      for (int f = lane; f < 64; f += 32) At[ke * LDA + f] = (_Float16)0.0f;
    }
  }
  __syncthreads();

  int hi = lane >> 4, n15 = lane & 15;
  for (int t = wv; t < MT * 4; t += 8) {
    int mi = t >> 2, ni = t & 3;
    v8f acc = {};
    v16h a0 = packA(At, mi, 0, lane);
    v16h bm0 = packB(Wl, 0, ni, lane);
    acc = __builtin_amdgcn_wmma_f32_16x16x32_f16(false, a0, false, bm0, (short)0, acc, false, false);
    v16h a1 = packA(At, mi, 1, lane);
    v16h bm1 = packB(Wl, 1, ni, lane);
    acc = __builtin_amdgcn_wmma_f32_16x16x32_f16(false, a1, false, bm1, (short)0, acc, false, false);
    int n = ni * 16 + n15;
    float bb = B1s[n];
    float wn = W2s[n];
#pragma unroll
    for (int r = 0; r < 8; ++r) {
      int row = mi * 16 + r + hi * 8;
      float val = siluf(acc[r] + bb) * wn;
      val = (row < NPB) ? val : 0.0f;
      // reduce across the 16 lanes of this half (columns of this N tile)
      for (int off = 1; off < 16; off <<= 1) val += __shfl_xor(val, off, 32);
      if (n15 == 0) phiW[wv][row] += val;  // lanes 0 & 16 write distinct rows; serial per wave
    }
  }
  __syncthreads();

  float b2v = b2[0];
  for (int ke = tid; ke < NPB; ke += 256) {
    float s = 0.0f;
    for (int w = 0; w < 8; ++w) s += phiW[w][ke];
    float phi = s + b2v;
    int e = src * NPB + ke;
    contrib[ke * 3 + 0] = CD[e * 3 + 0] * phi;
    contrib[ke * 3 + 1] = CD[e * 3 + 1] * phi;
    contrib[ke * 3 + 2] = CD[e * 3 + 2] * phi;
  }
  __syncthreads();
  if (tid < 3) {
    float s = 0.0f;
    for (int k = 0; k < NPB; ++k) s += contrib[k * 3 + tid];
    X[src * 3 + tid] += s * 0.01f;
  }
}

// ---------------- finalize: emb_out, atom decoder, centered velocity, output --------------
__global__ void __launch_bounds__(1024)
finalize_kernel(const float* __restrict__ H, const float* __restrict__ X,
                const float* __restrict__ X0, const int* __restrict__ mask_a,
                const int* __restrict__ mask_r, const float* __restrict__ P,
                float* __restrict__ out) {
  __shared__ float velS[NTOT_ * 3];
  __shared__ float bmean[NB * 3];
  __shared__ int nanflag;
  int tid = threadIdx.x;
  if (tid == 0) nanflag = 0;
  __syncthreads();
  if (tid < NTOT_) {
    for (int c = 0; c < 3; ++c) {
      float v = X[tid * 3 + c] - X0[tid * 3 + c];
      velS[tid * 3 + c] = v;
      if (v != v) atomicOr(&nanflag, 1);
    }
  }
  __syncthreads();
  if (tid < NB * 3) {
    int bb = tid / 3, c = tid - bb * 3;
    float s = 0.0f; int cnt = 0;
    for (int i = 0; i < NTOT_; ++i) {
      int bi = (i < NTOTA) ? mask_a[i] : mask_r[i - NTOTA];
      if (bi == bb) { s += velS[i * 3 + c]; ++cnt; }
    }
    bmean[tid] = s / (float)cnt;
  }
  __syncthreads();
  if (tid < NTOTA) {
    const float* h = H + tid * 64;
    const float* Wo = P + OFF_EMBO_W;
    const float* bo = P + OFF_EMBO_B;
    float hf[16];
    for (int j = 0; j < 16; ++j) {
      float s = bo[j];
      for (int f = 0; f < 64; ++f) s += h[f] * Wo[f * 17 + j];
      hf[j] = s;
    }
    float a1[32];
    for (int o = 0; o < 32; ++o) {
      float s = P[OFF_AD0_B + o];
      for (int f = 0; f < 16; ++f) s += hf[f] * P[OFF_AD0_W + f * 32 + o];
      a1[o] = siluf(s);
    }
    float a2[16];
    for (int o = 0; o < 16; ++o) {
      float s = P[OFF_AD1_B + o];
      for (int f = 0; f < 32; ++f) s += a1[f] * P[OFF_AD1_W + f * 16 + o];
      a2[o] = siluf(s);
    }
    int bb = mask_a[tid];
    for (int c = 0; c < 3; ++c) {
      float s = P[OFF_AD2_B + c];
      for (int f = 0; f < 16; ++f) s += a2[f] * P[OFF_AD2_W + f * 3 + c];
      float v = nanflag ? 0.0f : (velS[tid * 3 + c] - bmean[bb * 3 + c]);
      out[tid * 3 + c] = v + s;
    }
  }
}

// ---------------- host orchestration ----------------
extern "C" void kernel_launch(void* const* d_in, const int* in_sizes, int n_in,
                              void* d_out, int out_size, void* d_ws, size_t ws_size,
                              hipStream_t stream) {
  (void)in_sizes; (void)n_in; (void)out_size;
  const float* xa  = (const float*)d_in[0];
  const float* xr  = (const float*)d_in[1];
  const float* tsc = (const float*)d_in[2];
  const float* P   = (const float*)d_in[3];
  const int* mask_a = (const int*)d_in[4];
  const int* mask_r = (const int*)d_in[5];
  float* out = (float*)d_out;

  float* ws = (float*)d_ws;
  size_t off = 0;
  float* H   = ws + off; off += (size_t)NTOT_ * HID;
  float* X   = ws + off; off += (size_t)NTOT_ * 3;
  float* X0  = ws + off; off += (size_t)NTOT_ * 3;
  float* H17 = ws + off; off += (size_t)NTOT_ * 17;
  float* D0  = ws + off; off += (size_t)EALL;
  float* DC  = ws + off; off += (size_t)EALL;
  float* CD  = ws + off; off += (size_t)EALL * 3;
  float* EA  = ws + off; off += (size_t)EAT * 16;
  float* P1  = ws + off; off += (size_t)NTOT_ * HID;
  float* P2  = ws + off; off += (size_t)NTOT_ * HID;
  float* AGG = ws + off; off += (size_t)NTOT_ * HID;
  float* T1  = ws + off; off += (size_t)NTOT_ * HID;
  if (ws_size < off * sizeof(float)) return;  // scratch too small; bail safely

  dim3 b256(256);
  int gNode = (NTOT_ + 255) / 256;
  int gNL = (NTOT_ * HID + 255) / 256;
  int gE = (EALL + 255) / 256;

  setup_nodes_kernel<<<gNode, b256, 0, stream>>>(xa, xr, tsc, P, H17, X, X0);
  node_linear_kernel<<<gNL, b256, 0, stream>>>(H17, 17, nullptr, 0, P + OFF_EMB_W, P + OFF_EMB_B,
                                               H, HID, NTOT_, 0, 0);
  ea16_kernel<<<(EAT + 255) / 256, b256, 0, stream>>>(EA);
  edge_geom_kernel<<<gE, b256, 0, stream>>>(X0, D0, nullptr);

  for (int l = 0; l < 4; ++l) {
    const float* blk = P + OFF_BLOCKS + (size_t)l * BLOCK_STRIDE;
    edge_geom_kernel<<<gE, b256, 0, stream>>>(X, DC, CD);
    for (int g = 0; g < 2; ++g) {
      const float* gb = blk + GCLS_OFF + (size_t)g * GCL_STRIDE;
      // src / dst projections through e0 weight sub-blocks
      node_linear_kernel<<<gNL, b256, 0, stream>>>(H, 64, nullptr, 0, gb, nullptr, P1, 64, NTOT_, 0, 0);
      node_linear_kernel<<<gNL, b256, 0, stream>>>(H, 64, nullptr, 0, gb + 64 * 64, nullptr, P2, 64, NTOT_, 0, 0);
      gcl_edge_kernel<<<NTOT_, b256, 0, stream>>>(P1, P2, DC, D0, EA,
                                                  gb + 128 * 64, gb + 9344, gb + 9408, gb + 13504, AGG);
      // node update: h += n1( silu( n0([h, agg]) ) )
      node_linear_kernel<<<gNL, b256, 0, stream>>>(H, 64, AGG, 64, gb + 13568, gb + 21760, T1, 64, NTOT_, 1, 0);
      node_linear_kernel<<<gNL, b256, 0, stream>>>(T1, 64, nullptr, 0, gb + 21824, gb + 25920, H, 64, NTOT_, 0, 1);
    }
    // coord update projections through coord0 weight sub-blocks
    node_linear_kernel<<<gNL, b256, 0, stream>>>(H, 64, nullptr, 0, blk, nullptr, P1, 64, NTOT_, 0, 0);
    node_linear_kernel<<<gNL, b256, 0, stream>>>(H, 64, nullptr, 0, blk + 64 * 64, nullptr, P2, 64, NTOT_, 0, 0);
    coord_edge_kernel<<<NTOT_, b256, 0, stream>>>(P1, P2, DC, D0, EA, CD,
                                                   blk + 128 * 64, blk + 9344, blk + 9408, blk + 13504,
                                                   blk + 13568, blk + 13632, X);
  }

  finalize_kernel<<<1, 1024, 0, stream>>>(H, X, X0, mask_a, mask_r, P, out);
}
